// RNNLM_65481071395040
// MI455X (gfx1250) — matrix operand order repaired and checked by reading, not compiled, not run on hardware
//
#include <hip/hip_runtime.h>
#include <math.h>

// ---------------------------------------------------------------------------
// RNN LM: B=8, N=512, V=8000, H=256 on MI455X (gfx1250, wave32, WMMA).
//
// Algebraic restructure: x is only consumed as x_t @ wx, so fold:
//   W2    = emb_w @ wx                (8000x256, parallel, f16 WMMA)
//   bias2 = emb_b @ wx + hb
//   xw    = inputs @ W2 + bias2       (HBM-bound: 131MB -> ~5.6us floor)
//   scan: h = tanh(h @ wh + xw_t)     (8-deep dependent WMMA chain/step)
//   logits = h @ out_w + out_b ; softmax
// ---------------------------------------------------------------------------

typedef __attribute__((ext_vector_type(16))) _Float16 v16h;
typedef __attribute__((ext_vector_type(8)))  _Float16 v8h;
typedef __attribute__((ext_vector_type(8)))  float    v8f;
typedef __attribute__((ext_vector_type(4)))  float    v4f;

union V16U { v16h v; v8h h[2]; };

#define WMMA_F16(a, b, c) \
  __builtin_amdgcn_wmma_f32_16x16x32_f16(false, (a), false, (b), (short)0, (c), false, false)

static constexpr int BATCH = 8;
static constexpr int SEQ   = 512;
static constexpr int VOC   = 8000;
static constexpr int HID   = 256;
static constexpr int MROWS = BATCH * SEQ; // 4096

// ---------------------------------------------------------------------------
// Transpose f32 (R x C) -> f16 (C x R).
// ---------------------------------------------------------------------------
__global__ __launch_bounds__(256) void k_transpose(const float* __restrict__ in,
                                                   _Float16* __restrict__ out,
                                                   int R, int C) {
  int idx = blockIdx.x * 256 + threadIdx.x;
  if (idx >= R * C) return;
  int r = idx / C, c = idx % C;
  out[(size_t)c * R + r] = (_Float16)in[idx];
}

// ---------------------------------------------------------------------------
// bias2[n] = sum_h emb_b[h]*wx[h,n] + hb[n]
// ---------------------------------------------------------------------------
__global__ __launch_bounds__(256) void k_bias2(const float* __restrict__ emb_b,
                                               const float* __restrict__ wx,
                                               const float* __restrict__ hb,
                                               float* __restrict__ bias2) {
  const int n = threadIdx.x;
  float s = hb[n];
  for (int h = 0; h < HID; ++h) s += emb_b[h] * wx[h * HID + n];
  bias2[n] = s;
}

// ---------------------------------------------------------------------------
// W2T(256 x 8000) : W2T[n,v] = sum_h wx[h,n]*emb_w[v,h]
// A = wxT (f16, K-contig), B^T = emb_w itself (f32 rows, cvt in-register).
// grid (125, 16), 4 waves/block, 1 voc-tile per wave.
// ---------------------------------------------------------------------------
__global__ __launch_bounds__(128) void k_w2(const _Float16* __restrict__ wxT,
                                            const float* __restrict__ emb_w,
                                            _Float16* __restrict__ w2T) {
  const int wave = threadIdx.x >> 5, lane = threadIdx.x & 31;
  const int r = lane & 15, hi = lane >> 4;
  const int n0 = (blockIdx.x * 4 + wave) * 16;  // voc
  const int m0 = blockIdx.y * 16;               // hid

  v8f acc = {};
#pragma unroll
  for (int k0 = 0; k0 < HID; k0 += 32) {
    const _Float16* qa = wxT + (m0 + r) * HID + k0 + hi * 16;
    V16U a;
    a.h[0] = *(const v8h*)qa;
    a.h[1] = *(const v8h*)(qa + 8);
    const float* qb = emb_w + (size_t)(n0 + r) * HID + k0 + hi * 16;
    V16U b;
    {
      v4f b0 = *(const v4f*)(qb);
      v4f b1 = *(const v4f*)(qb + 4);
      v4f b2 = *(const v4f*)(qb + 8);
      v4f b3 = *(const v4f*)(qb + 12);
#pragma unroll
      for (int i = 0; i < 4; ++i) {
        b.v[i]      = (_Float16)b0[i];
        b.v[4 + i]  = (_Float16)b1[i];
        b.v[8 + i]  = (_Float16)b2[i];
        b.v[12 + i] = (_Float16)b3[i];
      }
    }
    acc = WMMA_F16(a.v, b.v, acc);
  }
#pragma unroll
  for (int j = 0; j < 8; ++j)
    w2T[(size_t)(m0 + j + 8 * hi) * VOC + n0 + r] = (_Float16)acc[j];
}

// ---------------------------------------------------------------------------
// xw = inputs(4096x8000) @ W2(8000x256) + bias2 -> xw16 (f16).
// HBM-bound: 1 n-tile per wave => 4096 waves for latency hiding.
// grid = 1024 blocks x 4 waves; block covers (m-tile, n-quarter).
// ---------------------------------------------------------------------------
__global__ __launch_bounds__(128) void k_embed(const float* __restrict__ inp,
                                               const _Float16* __restrict__ w2T,
                                               const float* __restrict__ bias2,
                                               _Float16* __restrict__ xw16) {
  const int wave = threadIdx.x >> 5, lane = threadIdx.x & 31;
  const int r = lane & 15, hi = lane >> 4;
  const int m0 = (blockIdx.x >> 2) * 16;
  const int n  = ((blockIdx.x & 3) * 4 + wave) * 16 + r;
  const float* arow = inp + (size_t)(m0 + r) * VOC + hi * 8;
  const _Float16* brow = w2T + (size_t)n * VOC + hi * 8;

  v8f acc = {};
  for (int k0 = 0; k0 < VOC; k0 += 32) {
    __builtin_prefetch(arow + k0 + 512, 0, 3);  // global_prefetch_b8
    V16U a;
    {
      v4f a0 = *(const v4f*)(arow + k0);
      v4f a1 = *(const v4f*)(arow + k0 + 4);
      v4f a2 = *(const v4f*)(arow + k0 + 16);
      v4f a3 = *(const v4f*)(arow + k0 + 20);
#pragma unroll
      for (int i = 0; i < 4; ++i) {
        a.v[i]      = (_Float16)a0[i];
        a.v[4 + i]  = (_Float16)a1[i];
        a.v[8 + i]  = (_Float16)a2[i];
        a.v[12 + i] = (_Float16)a3[i];
      }
    }
    V16U b;
    b.h[0] = *(const v8h*)(brow + k0);
    b.h[1] = *(const v8h*)(brow + k0 + 16);
    acc = WMMA_F16(a.v, b.v, acc);
  }

  const float bias = bias2[n];
#pragma unroll
  for (int j = 0; j < 8; ++j)
    xw16[(size_t)(m0 + j + 8 * hi) * HID + n] = (_Float16)(acc[j] + bias);
}

// ---------------------------------------------------------------------------
// Scan: h = tanh(h @ wh + xw_t), 512 steps. Single workgroup, 16 waves,
// wave w owns columns [16w,16w+16). h double-buffered in LDS (16x256 f16;
// pad rows 8..15 are self-contained: row m of A only feeds row m of D).
// xw_t addend loads hoisted ahead of the 8-deep WMMA chain.
// ---------------------------------------------------------------------------
__global__ __launch_bounds__(512) void k_rnn(const _Float16* __restrict__ xw16,
                                             const _Float16* __restrict__ whT,
                                             _Float16* __restrict__ hT) {
  __shared__ _Float16 hbuf[2][16][HID];
  const int tid  = threadIdx.x;
  const int wave = tid >> 5, lane = tid & 31;
  const int r = lane & 15, hi = lane >> 4;
  const int n = wave * 16 + r;

  for (int i = tid; i < 16 * HID; i += 512) ((_Float16*)hbuf[0])[i] = (_Float16)0.f;
  __syncthreads();

  int cur = 0;
  for (int t = 0; t < SEQ; ++t) {
    float xv[8];
#pragma unroll
    for (int j = 0; j < 8; ++j) {
      const int m = j + 8 * hi;
      xv[j] = (m < BATCH) ? (float)xw16[(size_t)(m * SEQ + t) * HID + n] : 0.f;
    }

    v8f acc = {};
#pragma unroll
    for (int k0 = 0; k0 < HID; k0 += 32) {
      const _Float16* qa = &hbuf[cur][r][k0 + hi * 8];
      V16U a;
      a.h[0] = *(const v8h*)qa;
      a.h[1] = *(const v8h*)(qa + 16);
      const _Float16* qb = whT + n * HID + k0 + hi * 8;
      V16U b;
      b.h[0] = *(const v8h*)qb;
      b.h[1] = *(const v8h*)(qb + 16);
      acc = WMMA_F16(a.v, b.v, acc);
    }

    const int nxt = cur ^ 1;
#pragma unroll
    for (int j = 0; j < 8; ++j)
      hbuf[nxt][j + 8 * hi][n] = (_Float16)tanhf(acc[j] + xv[j]);
    __syncthreads();
    cur = nxt;
  }

  for (int i = tid; i < 16 * HID; i += 512) hT[i] = ((_Float16*)hbuf[cur])[i];
}

// ---------------------------------------------------------------------------
// logits = hT(16x256) @ out_w + out_b.  hT staged to LDS with CDNA5 async
// global->LDS copies (ASYNCcnt), then A-frags come from LDS.
// ---------------------------------------------------------------------------
__global__ __launch_bounds__(128) void k_logits(const _Float16* __restrict__ hT,
                                                const _Float16* __restrict__ outT,
                                                const float* __restrict__ out_b,
                                                float* __restrict__ logits) {
  __shared__ _Float16 hLds[16 * HID];  // 8KB
  const int tid  = threadIdx.x;
  const int wave = tid >> 5, lane = tid & 31;
  const int r = lane & 15, hi = lane >> 4;

  {
    unsigned int lbase = (unsigned int)(uintptr_t)hLds + (unsigned int)tid * 16u;
    const _Float16* g = hT + tid * 8;
#pragma unroll
    for (int it = 0; it < 4; ++it) {
      asm volatile("global_load_async_to_lds_b128 %0, %1, off"
                   :: "v"(lbase + (unsigned int)(it * 2048)),
                      "v"((const void*)(g + it * 1024))
                   : "memory");
    }
    asm volatile("s_wait_asynccnt 0x0" ::: "memory");
  }
  __syncthreads();

  const int n0 = (blockIdx.x * 4 + wave) * 16;
  v8f acc = {};
#pragma unroll
  for (int k0 = 0; k0 < HID; k0 += 32) {
    const _Float16* qa = hLds + r * HID + k0 + hi * 8;
    V16U a;
    a.h[0] = *(const v8h*)qa;
    a.h[1] = *(const v8h*)(qa + 16);
    const _Float16* qb = outT + (size_t)(n0 + r) * HID + k0 + hi * 8;
    V16U b;
    b.h[0] = *(const v8h*)qb;
    b.h[1] = *(const v8h*)(qb + 16);
    acc = WMMA_F16(a.v, b.v, acc);
  }

  if (hi == 0) {  // lanes 0..15 hold M=0..7 in VGPRs 0..7
    const int nn = n0 + r;
    const float bias = out_b[nn];
#pragma unroll
    for (int j = 0; j < 8; ++j)
      logits[(size_t)j * VOC + nn] = acc[j] + bias;
  }
}

// ---------------------------------------------------------------------------
// Row softmax over V=8000; one block per batch row.
// ---------------------------------------------------------------------------
__global__ __launch_bounds__(256) void k_softmax(const float* __restrict__ logits,
                                                 float* __restrict__ out) {
  __shared__ float red[256];
  const int b = blockIdx.x, tid = threadIdx.x;
  const float* row = logits + (size_t)b * VOC;

  float m = -3.4e38f;
  for (int i = tid; i < VOC; i += 256) m = fmaxf(m, row[i]);
  red[tid] = m;
  __syncthreads();
  for (int s = 128; s > 0; s >>= 1) {
    if (tid < s) red[tid] = fmaxf(red[tid], red[tid + s]);
    __syncthreads();
  }
  m = red[0];
  __syncthreads();

  float sum = 0.f;
  for (int i = tid; i < VOC; i += 256) sum += __expf(row[i] - m);
  red[tid] = sum;
  __syncthreads();
  for (int s = 128; s > 0; s >>= 1) {
    if (tid < s) red[tid] += red[tid + s];
    __syncthreads();
  }
  const float inv = 1.f / red[0];
  for (int i = tid; i < VOC; i += 256) out[(size_t)b * VOC + i] = __expf(row[i] - m) * inv;
}

// ---------------------------------------------------------------------------
// Launch. Workspace (~10.7 MB, 16B-aligned chunks):
//   w2T f16 256x8000 | outT f16 8000x256 | whT,wxT f16 256x256
//   xw16 f16 4096x256 | hT f16 16x256 | bias2 f32 256 | logits f32 8x8000
// ---------------------------------------------------------------------------
extern "C" void kernel_launch(void* const* d_in, const int* in_sizes, int n_in,
                              void* d_out, int out_size, void* d_ws, size_t ws_size,
                              hipStream_t stream) {
  const float* inp   = (const float*)d_in[0];
  const float* emb_w = (const float*)d_in[1];
  const float* emb_b = (const float*)d_in[2];
  const float* wx    = (const float*)d_in[3];
  const float* wh    = (const float*)d_in[4];
  const float* hb    = (const float*)d_in[5];
  const float* out_w = (const float*)d_in[6];
  const float* out_b = (const float*)d_in[7];
  float* out = (float*)d_out;

  char* p = (char*)d_ws;
  _Float16* w2T  = (_Float16*)p; p += (size_t)HID * VOC * 2;
  _Float16* outT = (_Float16*)p; p += (size_t)VOC * HID * 2;
  _Float16* whT  = (_Float16*)p; p += (size_t)HID * HID * 2;
  _Float16* wxT  = (_Float16*)p; p += (size_t)HID * HID * 2;
  _Float16* xw16 = (_Float16*)p; p += (size_t)MROWS * HID * 2;
  _Float16* hT   = (_Float16*)p; p += (size_t)16 * HID * 2;
  float* bias2   = (float*)p;    p += (size_t)HID * 4;
  float* logits  = (float*)p;    p += (size_t)BATCH * VOC * 4;

  k_transpose<<<(HID * VOC + 255) / 256, 256, 0, stream>>>(out_w, outT, HID, VOC);
  k_transpose<<<(HID * HID + 255) / 256, 256, 0, stream>>>(wh, whT, HID, HID);
  k_transpose<<<(HID * HID + 255) / 256, 256, 0, stream>>>(wx, wxT, HID, HID);
  k_bias2<<<1, 256, 0, stream>>>(emb_b, wx, hb, bias2);

  k_w2<<<dim3(125, 16), 128, 0, stream>>>(wxT, emb_w, w2T);
  k_embed<<<(MROWS / 16) * 4, 128, 0, stream>>>(inp, w2T, bias2, xw16);
  k_rnn<<<1, 512, 0, stream>>>(xw16, whT, hT);
  k_logits<<<VOC / 64, 128, 0, stream>>>(hT, outT, out_b, logits);
  k_softmax<<<BATCH, 256, 0, stream>>>(logits, out);
}
